// PoseSplatter_24481313587944
// MI455X (gfx1250) — compile-verified, matrix operand order checked by reading, not run on hardware
//
#include <hip/hip_runtime.h>

#define G3      262144
#define NSEL    2048
#define HIMG    256
#define WIMG    256
#define NVIEW   6
#define VOXSZ   (0.18f/64.0f)
#define NEARP   0.01f

typedef __attribute__((ext_vector_type(16))) _Float16     v16h;
typedef __attribute__((ext_vector_type(8)))  float        v8f;
typedef __attribute__((ext_vector_type(4)))  unsigned int v4u;
typedef __attribute__((ext_vector_type(8)))  int          v8i_t;
typedef __attribute__((ext_vector_type(4)))  int          v4i_t;

// ---------------------------------------------------------------------------
// TDM: 1-D tile load Global -> LDS (D# built per CDNA5 ISA)
// group0: [1:0]=count=1, [63:32]=lds_addr, [120:64]=global_addr, [127:126]=type(2)
// group1: data_size=2(4B) @ [17:16]; tensor_dim0 @ [79:48]; tensor_dim1=1 @[111:80];
//         tile_dim0 @ [127:112]; tile_dim1=0 (1-D); tensor_dim0_stride @ [207:160]
// 6-arg builtin (clang-23 lane): (g0, g1, g2, g3, g4, cpol)
// ---------------------------------------------------------------------------
__device__ __forceinline__ void tdm_load_1d(unsigned lds_off, const void* gptr,
                                            unsigned ndwords) {
  unsigned long long ga = (unsigned long long)(size_t)gptr;
  v4u g0;
  g0[0] = 1u;                                            // count=1
  g0[1] = lds_off;                                       // lds_addr
  g0[2] = (unsigned)(ga & 0xFFFFFFFFull);                // global_addr lo
  g0[3] = ((unsigned)(ga >> 32) & 0x01FFFFFFu) | 0x80000000u; // addr hi | type=2
  v8i_t g1;
  g1[0] = (int)(2u << 16);                               // data_size = 4 bytes
  g1[1] = (int)((ndwords & 0xFFFFu) << 16);              // tensor_dim0[15:0]
  g1[2] = (int)(((ndwords >> 16) & 0xFFFFu) | (1u << 16)); // dim0 hi | tensor_dim1=1
  g1[3] = (int)((ndwords & 0xFFFFu) << 16);              // tile_dim0
  g1[4] = 0;                                             // tile_dim1=0 (1-D), tile_dim2=0
  g1[5] = (int)ndwords;                                  // tensor_dim0_stride lo
  g1[6] = 0;
  g1[7] = 0;
  v4i_t zz4 = {0, 0, 0, 0};
  v8i_t zz8 = {0, 0, 0, 0, 0, 0, 0, 0};
  __builtin_amdgcn_tensor_load_to_lds(g0, g1, zz4, zz4, zz8, 0);
}

__device__ __forceinline__ float sigm(float x) { return 1.0f / (1.0f + __expf(-x)); }

// ---------------------------------------------------------------------------
// K1: per-voxel multi-view sampling -> features + level histogram per block
// ---------------------------------------------------------------------------
__global__ __launch_bounds__(256) void k_voxel(
    const float* __restrict__ mask, const float* __restrict__ img,
    const float* __restrict__ p3d, const float* __restrict__ angle,
    const float* __restrict__ grid, const float* __restrict__ Ks,
    const float* __restrict__ vms,
    float4* __restrict__ voxfeat, int* __restrict__ level, int* __restrict__ bhist) {
  __shared__ int lh[8];
  int t = threadIdx.x, b = blockIdx.x;
  if (t < 8) lh[t] = 0;
  __syncthreads();
  int v = b * 256 + t;
  float a = angle[0];
  float ca = __cosf(a), sa = __sinf(a);
  float gx = grid[v*3+0], gy = grid[v*3+1], gz = grid[v*3+2];
  float pwx = ca*gx - sa*gy + p3d[0];
  float pwy = sa*gx + ca*gy + p3d[1];
  float pwz = gz + p3d[2];
  int cnt = 0; float cr = 0.f, cg = 0.f, cb = 0.f;
  #pragma unroll
  for (int c = 0; c < NVIEW; ++c) {
    const float* R = vms + c*16;
    float X = R[0]*pwx + R[1]*pwy + R[2]*pwz  + R[3];
    float Y = R[4]*pwx + R[5]*pwy + R[6]*pwz  + R[7];
    float Z = R[8]*pwx + R[9]*pwy + R[10]*pwz + R[11];
    const float* Kc = Ks + c*9;
    float iz = 1.0f / Z;
    float u  = (Kc[0]*X + Kc[2]*Z) * iz;
    float vv = (Kc[4]*Y + Kc[5]*Z) * iz;
    int ui = (int)rintf(u);  ui = min(max(ui, 0), WIMG-1);
    int vi = (int)rintf(vv); vi = min(max(vi, 0), HIMG-1);
    bool inb = (u >= 0.f) && (u < (float)WIMG) && (vv >= 0.f) && (vv < (float)HIMG) && (Z > NEARP);
    if (inb && mask[(c*HIMG + vi)*WIMG + ui] > 0.5f) {
      const float* ip = img + (((c*HIMG) + vi)*WIMG + ui) * 3;
      cnt++; cr += ip[0]; cg += ip[1]; cb += ip[2];
    }
  }
  float frac = (float)cnt * (1.0f/6.0f);
  float inv  = 1.0f / (float)max(cnt, 1);
  float c0, c1, c2;
  if (cnt == 6) { c0 = cr*inv; c1 = cg*inv; c2 = cb*inv; }
  else          { c0 = c1 = c2 = 0.45f; }
  voxfeat[v] = make_float4(10.f*frac - 5.f, c0, c1, c2);
  level[v] = cnt;
  atomicAdd(&lh[cnt], 1);
  __syncthreads();
  if (t < 8) bhist[b*8 + t] = lh[t];
}

// ---------------------------------------------------------------------------
// K2: exact top-k via counting sort over the 7 discrete prob levels
// ---------------------------------------------------------------------------
__global__ void k_scan(const int* __restrict__ bhist, int* __restrict__ boff) {
  if (threadIdx.x != 0 || blockIdx.x != 0) return;
  int total[7];
  for (int L = 0; L < 7; ++L) total[L] = 0;
  for (int b = 0; b < 1024; ++b)
    for (int L = 0; L < 7; ++L) total[L] += bhist[b*8 + L];
  int run[7]; int base = 0;
  for (int L = 6; L >= 0; --L) { run[L] = base; base += total[L]; }
  for (int b = 0; b < 1024; ++b)
    for (int L = 0; L < 7; ++L) { boff[b*8 + L] = run[L]; run[L] += bhist[b*8 + L]; }
}

__global__ __launch_bounds__(256) void k_select(
    const int* __restrict__ level, const float4* __restrict__ voxfeat,
    const int* __restrict__ boff, int* __restrict__ selidx,
    float* __restrict__ selprob, _Float16* __restrict__ feats16) {
  __shared__ int wcnt[8][8];
  int t = threadIdx.x, b = blockIdx.x;
  int lane = t & 31, wave = t >> 5;
  int v = b*256 + t;
  int L = level[v];
  int wr = 0;
  #pragma unroll
  for (int Lq = 0; Lq < 7; ++Lq) {
    unsigned bal = (unsigned)__ballot(L == Lq);
    if (L == Lq) wr = __popc(bal & ((1u << lane) - 1u));
    if (lane == 0) wcnt[wave][Lq] = __popc(bal);
  }
  __syncthreads();
  int woff = 0;
  for (int ww = 0; ww < wave; ++ww) woff += wcnt[ww][L];
  int rank = boff[b*8 + L] + woff + wr;
  if (rank < NSEL) {
    selidx[rank] = v;
    float4 f = voxfeat[v];
    selprob[rank] = sigm(f.x - 0.25f);
    _Float16* fp = feats16 + rank*32;
    fp[0] = (_Float16)f.x; fp[1] = (_Float16)f.y;
    fp[2] = (_Float16)f.z; fp[3] = (_Float16)f.w;
    #pragma unroll
    for (int i = 4; i < 32; ++i) fp[i] = (_Float16)0.f;
  }
}

// ---------------------------------------------------------------------------
// Weight prep: pad K 8->32 and N 14->16, convert to f16 for WMMA
// ---------------------------------------------------------------------------
__global__ void k_prepw(const float* __restrict__ w1, const float* __restrict__ w2,
                        _Float16* __restrict__ w1p, _Float16* __restrict__ w2p) {
  int i = blockIdx.x*256 + threadIdx.x;
  if (i < 32*128) {
    int k = i >> 7, n = i & 127;
    w1p[i] = (_Float16)(k < 8 ? w1[k*128 + n] : 0.f);
  } else if (i < 32*128 + 128*16) {
    int j = i - 32*128;
    int k = j >> 4, n = j & 15;
    w2p[j] = (_Float16)(n < 14 ? w2[k*14 + n] : 0.f);
  }
}

// Per-lane K index for 16-bit A/B tiles (ISA 7.12.2 16x32 layout)
__device__ __forceinline__ int kidx(int h, int lhi) {
  return ((h & 8) << 1) + (lhi << 3) + (h & 7);
}

// ---------------------------------------------------------------------------
// MLP layer 1: (2048x32) @ (32x128) via v_wmma_f32_16x16x32_f16, fused ReLU
// block = 8 waves; wave w -> N-tile w, block b -> M-tile b
// ---------------------------------------------------------------------------
__global__ __launch_bounds__(256) void k_mlp1(
    const _Float16* __restrict__ feats16, const _Float16* __restrict__ w1p,
    const float* __restrict__ b1, _Float16* __restrict__ h16) {
  int lane = threadIdx.x & 31, wave = threadIdx.x >> 5;
  int mt = blockIdx.x, nt = wave;
  int lhi = lane >> 4;
  int mrow = mt*16 + (lane & 15);
  int ncol = nt*16 + (lane & 15);
  v16h A, B;
  #pragma unroll
  for (int h = 0; h < 16; ++h) {
    int K = kidx(h, lhi);
    A[h] = feats16[mrow*32 + K];
    B[h] = w1p[K*128 + ncol];
  }
  v8f acc = {};
  acc = __builtin_amdgcn_wmma_f32_16x16x32_f16(false, A, false, B,
                                               (short)0, acc, false, false);
  float bias = b1[ncol];
  #pragma unroll
  for (int r = 0; r < 8; ++r) {
    int row = mt*16 + r + (lhi << 3);
    h16[row*128 + ncol] = (_Float16)fmaxf(acc[r] + bias, 0.f);
  }
}

// ---------------------------------------------------------------------------
// MLP layer 2: (2048x128) @ (128x16) -- 4 accumulating WMMA k-steps
// ---------------------------------------------------------------------------
__global__ __launch_bounds__(256) void k_mlp2(
    const _Float16* __restrict__ h16, const _Float16* __restrict__ w2p,
    const float* __restrict__ b2, float* __restrict__ mlpout) {
  int lane = threadIdx.x & 31, wave = threadIdx.x >> 5;
  int mt = blockIdx.x*8 + wave;
  int lhi = lane >> 4;
  int mrow = mt*16 + (lane & 15);
  int ncol = lane & 15;
  v8f acc = {};
  #pragma unroll
  for (int kk = 0; kk < 4; ++kk) {
    v16h A, B;
    #pragma unroll
    for (int h = 0; h < 16; ++h) {
      int K = kk*32 + kidx(h, lhi);
      A[h] = h16[mrow*128 + K];
      B[h] = w2p[K*16 + ncol];
    }
    acc = __builtin_amdgcn_wmma_f32_16x16x32_f16(false, A, false, B,
                                                 (short)0, acc, false, false);
  }
  #pragma unroll
  for (int r = 0; r < 8; ++r) {
    int row = mt*16 + r + (lhi << 3);
    if (ncol < 14) mlpout[row*14 + ncol] = acc[r] + b2[ncol];
  }
}

// ---------------------------------------------------------------------------
// K6: per-gaussian param decode, quat<->mat round-trip (Jacobi eigh),
//     2D covariance projection, depth key
// ---------------------------------------------------------------------------
__global__ __launch_bounds__(256) void k_gauss(
    const float* __restrict__ grid, const float* __restrict__ p3d,
    const float* __restrict__ angle, const float* __restrict__ Ks,
    const float* __restrict__ vms, const int* __restrict__ viewnum,
    const float* __restrict__ scale_param,
    const float* __restrict__ mlpout, const int* __restrict__ selidx,
    const float* __restrict__ selprob,
    float* __restrict__ rec, unsigned long long* __restrict__ zkey) {
  int r = blockIdx.x*256 + threadIdx.x;
  if (r >= NSEL) return;
  const float* o = mlpout + r*14;
  float q0 = o[0], q1 = o[1], q2 = o[2], q3 = o[3];
  float sp = scale_param[0];
  float s0 = __expf(o[4]+sp), s1 = __expf(o[5]+sp), s2 = __expf(o[6]+sp);
  float col0 = fminf(sigm(o[8]),  0.99f);
  float col1 = fminf(sigm(o[9]),  0.99f);
  float col2 = fminf(sigm(o[10]), 0.99f);
  float opac = sigm(fminf(fmaxf((selprob[r] - 0.25f) * (1.0f/0.75f), 0.f), 1.f));
  int v = selidx[r];
  float a = angle[0]; float ca = __cosf(a), sa = __sinf(a);
  float mx = grid[v*3+0] + 2.f*VOXSZ*tanhf(o[11]);
  float my = grid[v*3+1] + 2.f*VOXSZ*tanhf(o[12]);
  float mz = grid[v*3+2] + 2.f*VOXSZ*tanhf(o[13]);
  float wx = ca*mx - sa*my + p3d[0];
  float wy = sa*mx + ca*my + p3d[1];
  float wz = mz + p3d[2];

  // quat_to_mat4 (reference formula, 3x3 part)
  float M[3][3];
  float nq = q0*q0 + q1*q1 + q2*q2 + q3*q3;
  if (nq < 4.0f*1.1920929e-7f) {
    M[0][0]=1;M[0][1]=0;M[0][2]=0; M[1][0]=0;M[1][1]=1;M[1][2]=0; M[2][0]=0;M[2][1]=0;M[2][2]=1;
  } else {
    float sc = sqrtf(2.0f/nq);
    float u0=q0*sc,u1=q1*sc,u2=q2*sc,u3=q3*sc;
    float o11=u1*u1,o22=u2*u2,o33=u3*u3;
    float o12=u1*u2,o13=u1*u3,o23=u2*u3;
    float o10=u1*u0,o20=u2*u0,o30=u3*u0;
    M[0][0]=1.f-o22-o33; M[0][1]=o12-o30; M[0][2]=o13+o20;
    M[1][0]=o12-o30;     M[1][1]=1.f;     M[1][2]=o23-o10;
    M[2][0]=o13-o20;     M[2][1]=o23+o10; M[2][2]=1.f-o11-o22;
  }
  // r4 = Rz @ M
  float Rm[3][3];
  #pragma unroll
  for (int j = 0; j < 3; ++j) {
    Rm[0][j] = ca*M[0][j] - sa*M[1][j];
    Rm[1][j] = sa*M[0][j] + ca*M[1][j];
    Rm[2][j] = M[2][j];
  }
  float m00=Rm[0][0],m01=Rm[0][1],m02=Rm[0][2];
  float m10=Rm[1][0],m11=Rm[1][1],m12=Rm[1][2];
  float m20=Rm[2][0],m21=Rm[2][1],m22=Rm[2][2];
  float Am[4][4], Vm[4][4];
  Am[0][0]=m00-m11-m22; Am[0][1]=m01+m10; Am[0][2]=m02+m20; Am[0][3]=m21-m12;
  Am[1][1]=m11-m00-m22; Am[1][2]=m12+m21; Am[1][3]=m02-m20;
  Am[2][2]=m22-m00-m11; Am[2][3]=m10-m01;
  Am[3][3]=m00+m11+m22;
  Am[1][0]=Am[0][1]; Am[2][0]=Am[0][2]; Am[3][0]=Am[0][3];
  Am[2][1]=Am[1][2]; Am[3][1]=Am[1][3]; Am[3][2]=Am[2][3];
  #pragma unroll
  for (int i = 0; i < 4; ++i)
    #pragma unroll
    for (int j = 0; j < 4; ++j) { Am[i][j] *= (1.0f/3.0f); Vm[i][j] = (i==j)?1.f:0.f; }
  // Jacobi eigensolver, 8 fixed sweeps (fully unrolled -> registers only)
  const int PP[6] = {0,0,0,1,1,2}, QQ[6] = {1,2,3,2,3,3};
  #pragma unroll
  for (int sweep = 0; sweep < 8; ++sweep) {
    #pragma unroll
    for (int pi = 0; pi < 6; ++pi) {
      const int p = PP[pi], q = QQ[pi];
      float apq = Am[p][q];
      if (fabsf(apq) > 1e-12f) {
        float th  = (Am[q][q] - Am[p][p]) / (2.f*apq);
        float tt  = ((th >= 0.f) ? 1.f : -1.f) / (fabsf(th) + sqrtf(th*th + 1.f));
        float cth = rsqrtf(tt*tt + 1.f), sth = tt*cth;
        #pragma unroll
        for (int k2=0;k2<4;k2++){float akp=Am[k2][p],akq=Am[k2][q];
          Am[k2][p]=cth*akp-sth*akq; Am[k2][q]=sth*akp+cth*akq;}
        #pragma unroll
        for (int k2=0;k2<4;k2++){float apk=Am[p][k2],aqk=Am[q][k2];
          Am[p][k2]=cth*apk-sth*aqk; Am[q][k2]=sth*apk+cth*aqk;}
        #pragma unroll
        for (int k2=0;k2<4;k2++){float vkp=Vm[k2][p],vkq=Vm[k2][q];
          Vm[k2][p]=cth*vkp-sth*vkq; Vm[k2][q]=sth*vkp+cth*vkq;}
      }
    }
  }
  int jm = 0; float best = Am[0][0];
  #pragma unroll
  for (int j = 1; j < 4; ++j) if (Am[j][j] > best) { best = Am[j][j]; jm = j; }
  float e0=Vm[0][0], e1=Vm[1][0], e2=Vm[2][0], e3=Vm[3][0];
  #pragma unroll
  for (int j = 1; j < 4; ++j)
    if (j == jm) { e0=Vm[0][j]; e1=Vm[1][j]; e2=Vm[2][j]; e3=Vm[3][j]; }
  float Q0=e3, Q1=e0, Q2=e1, Q3=e2;            // reorder [3,0,1,2]
  if (Q0 < 0.f) { Q0=-Q0; Q1=-Q1; Q2=-Q2; Q3=-Q3; }
  // splat: normalized quat -> rotation
  float qn = sqrtf(Q0*Q0+Q1*Q1+Q2*Q2+Q3*Q3) + 1e-12f;
  float iw=Q0/qn, ix=Q1/qn, iy=Q2/qn, iz=Q3/qn;
  float Rq[3][3];
  Rq[0][0]=1.f-2.f*(iy*iy+iz*iz); Rq[0][1]=2.f*(ix*iy-iw*iz); Rq[0][2]=2.f*(ix*iz+iw*iy);
  Rq[1][0]=2.f*(ix*iy+iw*iz); Rq[1][1]=1.f-2.f*(ix*ix+iz*iz); Rq[1][2]=2.f*(iy*iz-iw*ix);
  Rq[2][0]=2.f*(ix*iz-iw*iy); Rq[2][1]=2.f*(iy*iz+iw*ix); Rq[2][2]=1.f-2.f*(ix*ix+iy*iy);
  float ss[3] = {s0*s0, s1*s1, s2*s2};
  float cov3[3][3];
  #pragma unroll
  for (int i = 0; i < 3; ++i)
    #pragma unroll
    for (int j = 0; j < 3; ++j) {
      float acc2 = 0.f;
      #pragma unroll
      for (int k2 = 0; k2 < 3; ++k2) acc2 += Rq[i][k2]*ss[k2]*Rq[j][k2];
      cov3[i][j] = acc2;
    }
  int vn = viewnum[0];
  const float* VM = vms + vn*16;
  float R3[3][3] = {{VM[0],VM[1],VM[2]},{VM[4],VM[5],VM[6]},{VM[8],VM[9],VM[10]}};
  float X = R3[0][0]*wx + R3[0][1]*wy + R3[0][2]*wz + VM[3];
  float Y = R3[1][0]*wx + R3[1][1]*wy + R3[1][2]*wz + VM[7];
  float Z = R3[2][0]*wx + R3[2][1]*wy + R3[2][2]*wz + VM[11];
  float tmp[3][3], covc[3][3];
  #pragma unroll
  for (int i = 0; i < 3; ++i)
    #pragma unroll
    for (int j = 0; j < 3; ++j) {
      float acc2 = 0.f;
      #pragma unroll
      for (int k2 = 0; k2 < 3; ++k2) acc2 += R3[i][k2]*cov3[k2][j];
      tmp[i][j] = acc2;
    }
  #pragma unroll
  for (int i = 0; i < 3; ++i)
    #pragma unroll
    for (int j = 0; j < 3; ++j) {
      float acc2 = 0.f;
      #pragma unroll
      for (int k2 = 0; k2 < 3; ++k2) acc2 += tmp[i][k2]*R3[j][k2];
      covc[i][j] = acc2;
    }
  const float* Kc = Ks + vn*9;
  float fx = Kc[0], cxp = Kc[2], fy = Kc[4], cyp = Kc[5];
  float rz = 1.f/Z;
  float J00 = fx*rz, J02 = -fx*X*rz*rz, J11 = fy*rz, J12 = -fy*Y*rz*rz;
  float c00 = J00*J00*covc[0][0] + 2.f*J00*J02*covc[0][2] + J02*J02*covc[2][2];
  float c01 = J00*(covc[0][1]*J11 + covc[0][2]*J12) + J02*(covc[2][1]*J11 + covc[2][2]*J12);
  float c11 = J11*J11*covc[1][1] + 2.f*J11*J12*covc[1][2] + J12*J12*covc[2][2];
  float a2 = c00 + 0.3f, b2v = c01, cc2 = c11 + 0.3f;
  float det   = a2*cc2 - b2v*b2v;
  float det_s = (det > 1e-12f) ? det : 1.f;
  float cA = cc2/det_s, cB = -b2v/det_s, cC = a2/det_s;
  float lam = 0.5f*(a2+cc2) + sqrtf(fmaxf(0.25f*(a2-cc2)*(a2-cc2) + b2v*b2v, 1e-12f));
  float radius = ceilf(3.f*sqrtf(lam));
  float mux = fx*X*rz + cxp, muy = fy*Y*rz + cyp;
  bool valid = (Z > NEARP) && (det > 1e-12f) && (radius > 2.0f);
  float op = valid ? opac : 0.f;
  float* rp = rec + r*12;
  rp[0]=mux; rp[1]=muy; rp[2]=cA; rp[3]=cB; rp[4]=cC; rp[5]=op;
  rp[6]=col0; rp[7]=col1; rp[8]=col2; rp[9]=Z; rp[10]=0.f; rp[11]=0.f;
  unsigned fb = __float_as_uint(Z);
  fb = (fb & 0x80000000u) ? ~fb : (fb | 0x80000000u);
  zkey[r] = ((unsigned long long)fb << 32) | (unsigned)r;   // stable z-ascending key
}

// ---------------------------------------------------------------------------
// K7: single-workgroup LDS bitonic sort of 2048 (z,idx) keys
// ---------------------------------------------------------------------------
__global__ __launch_bounds__(1024) void k_sort(const unsigned long long* __restrict__ zkey,
                                               int* __restrict__ perm) {
  __shared__ unsigned long long key[NSEL];
  int t = threadIdx.x;
  key[t]        = zkey[t];
  key[t + 1024] = zkey[t + 1024];
  __syncthreads();
  for (int k = 2; k <= NSEL; k <<= 1) {
    for (int j = k >> 1; j > 0; j >>= 1) {
      #pragma unroll
      for (int e = 0; e < 2; ++e) {
        int i = t + e*1024;
        int ixj = i ^ j;
        if (ixj > i) {
          bool asc = ((i & k) == 0);
          unsigned long long a = key[i], b = key[ixj];
          if ((a > b) == asc) { key[i] = b; key[ixj] = a; }
        }
      }
      __syncthreads();
    }
  }
  perm[t]        = (int)(unsigned)(key[t] & 0xFFFFFFFFull);
  perm[t + 1024] = (int)(unsigned)(key[t + 1024] & 0xFFFFFFFFull);
}

__global__ __launch_bounds__(256) void k_gather(const float* __restrict__ rec,
                                                const int* __restrict__ perm,
                                                float* __restrict__ gsort) {
  int r = blockIdx.x*256 + threadIdx.x;
  if (r >= NSEL) return;
  int p = perm[r];
  const float* s = rec + p*12;
  float* d = gsort + r*12;
  #pragma unroll
  for (int i = 0; i < 12; ++i) d[i] = s[i];
}

// ---------------------------------------------------------------------------
// K9: tile renderer. 16x16 pixel tile per block, 1 pixel per thread.
// Sorted gaussian table streamed Global->LDS with double-buffered TDM loads.
// ---------------------------------------------------------------------------
__global__ __launch_bounds__(256) void k_render(
    const float* __restrict__ gsort, const float* __restrict__ background,
    float* __restrict__ outp) {
  __shared__ float sg[2][3072];                      // 2 x 256 gaussians x 12 floats
  int t = threadIdx.x;
  int tilex = blockIdx.x & 15, tiley = blockIdx.x >> 4;
  float px = (float)(tilex*16 + (t & 15)) + 0.5f;
  float py = (float)(tiley*16 + (t >> 4)) + 0.5f;
  unsigned ldsbase = (unsigned)(size_t)(&sg[0][0]);  // addr[31:0] = LDS offset
  float T = 1.f, rr = 0.f, gg = 0.f, bb = 0.f;
  if (t < 32) tdm_load_1d(ldsbase, gsort, 3072);     // wave 0 drives the TDM
  for (int c = 0; c < 8; ++c) {
    if (t < 32) __builtin_amdgcn_s_wait_tensorcnt(0);
    __syncthreads();                                 // publish chunk c, fence prev compute
    if (t < 32 && c + 1 < 8)
      tdm_load_1d(ldsbase + (unsigned)((c+1) & 1) * 3072u * 4u,
                  gsort + (c+1)*3072, 3072);         // prefetch next chunk
    const float* gs = &sg[c & 1][0];
    for (int n = 0; n < 256; ++n) {
      const float* gp = gs + n*12;
      float dx = px - gp[0], dy = py - gp[1];
      float sig = 0.5f*(gp[2]*dx*dx + gp[4]*dy*dy) + gp[3]*dx*dy;
      float al  = gp[5] * __expf(-sig);
      if (sig >= 0.f && al >= (1.0f/255.0f)) {
        float aa = fminf(al, 0.999f);
        float w  = aa * T;
        rr += w*gp[6]; gg += w*gp[7]; bb += w*gp[8];
        T *= (1.f - aa);
      }
    }
  }
  int pix = (tiley*16 + (t >> 4))*WIMG + tilex*16 + (t & 15);
  float b0 = background[0], b1 = background[1], b2 = background[2];
  outp[pix*3+0] = fminf(fmaxf(rr + T*b0, 0.f), 1.f);
  outp[pix*3+1] = fminf(fmaxf(gg + T*b1, 0.f), 1.f);
  outp[pix*3+2] = fminf(fmaxf(bb + T*b2, 0.f), 1.f);
  outp[WIMG*HIMG*3 + pix] = 1.f - T;
}

// ---------------------------------------------------------------------------
extern "C" void kernel_launch(void* const* d_in, const int* in_sizes, int n_in,
                              void* d_out, int out_size, void* d_ws, size_t ws_size,
                              hipStream_t stream) {
  const float* mask   = (const float*)d_in[0];
  const float* img    = (const float*)d_in[1];
  const float* p3d    = (const float*)d_in[2];
  const float* angle  = (const float*)d_in[3];
  const float* grid   = (const float*)d_in[4];
  const float* Ks     = (const float*)d_in[5];
  const float* vms    = (const float*)d_in[6];
  const float* w1     = (const float*)d_in[7];
  const float* b1     = (const float*)d_in[8];
  const float* w2     = (const float*)d_in[9];
  const float* b2     = (const float*)d_in[10];
  const float* sparam = (const float*)d_in[11];
  const float* bg     = (const float*)d_in[12];
  const int*   vn     = (const int*)d_in[13];

  char* w = (char*)d_ws;
  float4*             voxfeat = (float4*)(w + 0);          // 4 MB
  int*                level   = (int*)(w + 4194304);       // 1 MB
  int*                bhist   = (int*)(w + 5242880);       // 32 KB
  int*                boff    = (int*)(w + 5275648);       // 32 KB
  int*                selidx  = (int*)(w + 5308416);       // 8 KB
  float*              selprob = (float*)(w + 5316608);     // 8 KB
  _Float16*           feats16 = (_Float16*)(w + 5324800);  // 128 KB
  _Float16*           w1p     = (_Float16*)(w + 5455872);  // 8 KB
  _Float16*           w2p     = (_Float16*)(w + 5464064);  // 4 KB
  _Float16*           h16     = (_Float16*)(w + 5468160);  // 512 KB
  float*              mlpout  = (float*)(w + 5992448);     // 112 KB
  float*              rec     = (float*)(w + 6107136);     // 96 KB
  unsigned long long* zkey    = (unsigned long long*)(w + 6205440); // 16 KB
  int*                perm    = (int*)(w + 6221824);       // 8 KB
  float*              gsort   = (float*)(w + 6230016);     // 96 KB
  float*              outp    = (float*)d_out;

  k_voxel <<<1024, 256, 0, stream>>>(mask, img, p3d, angle, grid, Ks, vms,
                                     voxfeat, level, bhist);
  k_scan  <<<1,    64,  0, stream>>>(bhist, boff);
  k_select<<<1024, 256, 0, stream>>>(level, voxfeat, boff, selidx, selprob, feats16);
  k_prepw <<<24,   256, 0, stream>>>(w1, w2, w1p, w2p);
  k_mlp1  <<<128,  256, 0, stream>>>(feats16, w1p, b1, h16);
  k_mlp2  <<<16,   256, 0, stream>>>(h16, w2p, b2, mlpout);
  k_gauss <<<8,    256, 0, stream>>>(grid, p3d, angle, Ks, vms, vn, sparam,
                                     mlpout, selidx, selprob, rec, zkey);
  k_sort  <<<1,   1024, 0, stream>>>(zkey, perm);
  k_gather<<<8,    256, 0, stream>>>(rec, perm, gsort);
  k_render<<<256,  256, 0, stream>>>(gsort, bg, outp);
  (void)in_sizes; (void)n_in; (void)out_size; (void)ws_size;
}
